// Multi_MST_GCN_86423331930150
// MI455X (gfx1250) — compile-verified, matrix operand order checked by reading, not run on hardware
//
#include <hip/hip_runtime.h>
#include <hip/hip_bf16.h>

// ---------------------------------------------------------------------------
// Types / helpers for CDNA5 WMMA (wave32, 16x16x32 bf16 -> f32)
// ---------------------------------------------------------------------------
typedef __attribute__((ext_vector_type(16))) __bf16 v16bf;
typedef __attribute__((ext_vector_type(8)))  float  v8f;

union FragU {
  v16bf v;
  uint4 q[2];
};

__device__ __forceinline__ unsigned short f2bf(float f) {
  unsigned int u = __float_as_uint(f);
  u += 0x7FFFu + ((u >> 16) & 1u);      // round-to-nearest-even
  return (unsigned short)(u >> 16);
}

__device__ __forceinline__ float sigm(float x) { return 1.0f / (1.0f + expf(-x)); }

// Fragment slot mapping for 16-bit 16x32 A / 32x16 B tiles (ISA 7.12.2):
//   K index c (0..31):  group g = c>>3 ; half = g&1 ; slot = (g>>1)*8 + (c&7)
//   lane = (row or col & 15) + 16*half ; fragment element j = slot.

// ---------------------------------------------------------------------------
// Generic GEMM: C[M,N] = act( (RS ? diag(rowScale) : I) * A[M,K] @ W[K,N]
//                             + (HASBIAS ? bias : 0) + (ADDC ? C : 0) )
// REQUIRES: M % 16 == 0, N % 64 == 0, K % 32 == 0, A/W 16-byte aligned.
// Block = 4 waves (128 thr); each wave owns one 16x16 C tile (block: 16 x 64).
// LDS ping-pong double buffering; tiles stored in WMMA *fragment order* so
// each lane reads its whole fragment with two 128-bit DS loads.
// ---------------------------------------------------------------------------
template <bool HASBIAS, bool RS, bool RELU, bool ADDC>
__global__ void __launch_bounds__(128)
gemm_bf16_wmma(const float* __restrict__ A, const float* __restrict__ W,
               const float* __restrict__ bias, const float* __restrict__ rowScale,
               float* __restrict__ C, int M, int K, int N) {
  // fragment-ordered tiles, row stride 24 ushorts (48B) to spread banks
  __shared__ unsigned short sAf[2][32][24];      // [buf][lane][slot]
  __shared__ unsigned short sBf[2][4][32][24];   // [buf][wave][lane][slot]

  const int row0 = blockIdx.x * 16;
  const int col0 = blockIdx.y * 64;
  const int lane = threadIdx.x;               // 0..31
  const int wv   = threadIdx.y;               // 0..3  (16-col sub-tile)
  const int tid  = wv * 32 + lane;

  // ---- per-thread staging roles (fixed across k-steps) ----
  // A: 16 rows x 32 cols ; 8 threads per row, float4 each
  const int ar  = tid >> 3;                   // 0..15
  const int ac4 = (tid & 7) * 4;              // 0,4,...,28
  const int ag    = ac4 >> 3;
  const int aslot = ((ag >> 1) << 3) + (ac4 & 7);
  const int alane = ar + ((ag & 1) << 4);
  // B: 32 rows(K) x 64 cols ; thread -> one col, half the K range, pairs along K
  const int bc      = tid & 63;               // 0..63
  const int bkh     = tid >> 6;               // 0 or 1 -> K base 0 / 16
  const int bwc     = bc >> 4;                // wave-col group
  const int blane15 = bc & 15;

  auto stage = [&](int buf, int k0) {
    // A tile (one float4 per thread, fragment-ordered store)
    {
      const float4 av = *(const float4*)(A + (size_t)(row0 + ar) * K + (k0 + ac4));
      float rs = 1.0f;
      if (RS) rs = rowScale[row0 + ar];
      ushort4 pk;
      pk.x = f2bf(av.x * rs);
      pk.y = f2bf(av.y * rs);
      pk.z = f2bf(av.z * rs);
      pk.w = f2bf(av.w * rs);
      *(ushort4*)&sAf[buf][alane][aslot] = pk;
    }
    // B tile (coalesced b32 loads, packed u32 fragment-ordered stores)
    {
      const float* Wc = W + (size_t)(k0 + bkh * 16) * N + (col0 + bc);
#pragma unroll
      for (int kk = 0; kk < 8; ++kk) {
        int kr = bkh * 16 + kk * 2;
        float v0 = Wc[(size_t)(kk * 2) * N];
        float v1 = Wc[(size_t)(kk * 2 + 1) * N];
        int g = kr >> 3;
        int half = g & 1;
        int slot = ((g >> 1) << 3) + (kr & 7);
        unsigned int pk = (unsigned int)f2bf(v0) | ((unsigned int)f2bf(v1) << 16);
        *(unsigned int*)&sBf[buf][bwc][blane15 + (half << 4)][slot] = pk;
      }
    }
  };

  v8f acc = {0.f, 0.f, 0.f, 0.f, 0.f, 0.f, 0.f, 0.f};

  const int nk = K >> 5;
  stage(0, 0);
  for (int ks = 0; ks < nk; ++ks) {
    const int p = ks & 1;
    __syncthreads();
    FragU a, b;
    a.q[0] = *(const uint4*)&sAf[p][lane][0];
    a.q[1] = *(const uint4*)&sAf[p][lane][8];
    b.q[0] = *(const uint4*)&sBf[p][wv][lane][0];
    b.q[1] = *(const uint4*)&sBf[p][wv][lane][8];
    if (ks + 1 < nk) stage(p ^ 1, (ks + 1) << 5);
    acc = __builtin_amdgcn_wmma_f32_16x16x32_bf16(
        false, a.v, false, b.v, (short)0, acc, false, false);
  }

  // ---- write back: C lane layout (lanes 0-15: M=r, 16-31: M=8+r; N=lane&15)
  const int mbase = row0 + (lane >> 4) * 8;
  const int gc    = col0 + wv * 16 + (lane & 15);
  float bv = 0.f;
  if (HASBIAS) bv = bias[gc];
  float cin[8];
  if (ADDC) {
#pragma unroll
    for (int r = 0; r < 8; ++r) cin[r] = C[(size_t)(mbase + r) * N + gc];
  }
#pragma unroll
  for (int r = 0; r < 8; ++r) {
    float v = acc[r] + bv;
    if (ADDC) v += cin[r];
    if (RELU) v = fmaxf(v, 0.f);
    C[(size_t)(mbase + r) * N + gc] = v;
  }
}

// ---------------------------------------------------------------------------
// Elementwise / graph kernels
// ---------------------------------------------------------------------------
__global__ void zero_kernel(float* __restrict__ p, size_t n) {
  size_t i = (size_t)blockIdx.x * blockDim.x + threadIdx.x;
  size_t s = (size_t)gridDim.x * blockDim.x;
  for (; i < n; i += s) p[i] = 0.f;
}

__global__ void count_kernel(const int* __restrict__ idx, float* __restrict__ cnt, int E) {
  int t = blockIdx.x * blockDim.x + threadIdx.x;
  if (t < E) atomicAdd(&cnt[idx[t]], 1.0f);
}

__global__ void norm_fin_kernel(float* __restrict__ c, int N) {
  int t = blockIdx.x * blockDim.x + threadIdx.x;
  if (t < N) c[t] = rsqrtf(fmaxf(c[t], 1.0f));
}

// agg[dst[e], :] += h[src[e], :] * nsrc[src[e]]   (32 lanes per edge, H cols)
__global__ void scatter_kernel(const float* __restrict__ h, const float* __restrict__ nsrc,
                               const int* __restrict__ src, const int* __restrict__ dst,
                               float* __restrict__ agg, int E, int H) {
  int t = blockIdx.x * blockDim.x + threadIdx.x;
  int e = t >> 5, lane = t & 31;
  if (e >= E) return;
  int s = src[e], d = dst[e];
  float ns = nsrc[s];
  const float* hs = h + (size_t)s * H;
  float* ad = agg + (size_t)d * H;
  for (int c = lane; c < H; c += 32) atomicAdd(&ad[c], hs[c] * ns);
}

// in-place: gp holds gate pre-activation; becomes fused = g*p1 + (1-g)*p2
__global__ void gatefuse_kernel(float* __restrict__ gp, const float* __restrict__ p1,
                                const float* __restrict__ p2, size_t n) {
  size_t i = (size_t)blockIdx.x * blockDim.x + threadIdx.x;
  if (i < n) {
    float g = sigm(gp[i]);
    gp[i] = g * p1[i] + (1.0f - g) * p2[i];
  }
}

// GRU combine + ReLU:  hout = relu((1-z)*n + z*h)
__global__ void gru_kernel(const float* __restrict__ gi, const float* __restrict__ gh,
                           const float* __restrict__ hfused, float* __restrict__ hout,
                           int N, int H) {
  size_t t = (size_t)blockIdx.x * blockDim.x + threadIdx.x;
  size_t total = (size_t)N * H;
  if (t >= total) return;
  size_t i = t / H, c = t % H;
  const float* giR = gi + i * (size_t)(3 * H);
  const float* ghR = gh + i * (size_t)(3 * H);
  float r = sigm(giR[c] + ghR[c]);
  float z = sigm(giR[H + c] + ghR[H + c]);
  float nn = tanhf(giR[2 * H + c] + r * ghR[2 * H + c]);
  float hh = hfused[i * (size_t)H + c];
  hout[i * (size_t)H + c] = fmaxf((1.0f - z) * nn + z * hh, 0.0f);
}

// rep[m] = [ e0[ei[m]] | e1[ei[m]] | c0[ci[m]] | c1[ci[m]] ]  (each H wide)
__global__ void gather_rep_kernel(const float* __restrict__ e0, const float* __restrict__ e1,
                                  const float* __restrict__ c0, const float* __restrict__ c1,
                                  const int* __restrict__ ei, const int* __restrict__ ci,
                                  float* __restrict__ rep, int M, int H) {
  int t = blockIdx.x * blockDim.x + threadIdx.x;
  int tot = M * 4 * H;
  if (t >= tot) return;
  int m = t / (4 * H), j = t % (4 * H);
  float v;
  if (j < H)            v = e0[(size_t)ei[m] * H + j];
  else if (j < 2 * H)   v = e1[(size_t)ei[m] * H + (j - H)];
  else if (j < 3 * H)   v = c0[(size_t)ci[m] * H + (j - 2 * H)];
  else                  v = c1[(size_t)ci[m] * H + (j - 3 * H)];
  rep[(size_t)m * 4 * H + j] = v;
}

__global__ void head_kernel(const float* __restrict__ x, const float* __restrict__ W3,
                            const float* __restrict__ b3, float* __restrict__ out,
                            int M, int K) {
  int m = blockIdx.x * blockDim.x + threadIdx.x;
  if (m >= M) return;
  float s = b3[0];
  const float* xr = x + (size_t)m * K;
  for (int k = 0; k < K; ++k) s += xr[k] * W3[k];
  out[m] = sigm(s);
}

// ---------------------------------------------------------------------------
// Host orchestration
// ---------------------------------------------------------------------------
extern "C" void kernel_launch(void* const* d_in, const int* in_sizes, int n_in,
                              void* d_out, int out_size, void* d_ws, size_t ws_size,
                              hipStream_t stream) {
  const float* feat_e = (const float*)d_in[0];
  const float* feat_o = (const float*)d_in[1];
  const float* feat_c = (const float*)d_in[2];
  const float* proj_e_w = (const float*)d_in[3];
  const float* proj_e_b = (const float*)d_in[4];
  const float* proj_o_w = (const float*)d_in[5];
  const float* proj_o_b = (const float*)d_in[6];
  const float* proj_c_w = (const float*)d_in[7];
  const float* proj_c_b = (const float*)d_in[8];
  const float* Wrel = (const float*)d_in[9];
  const float* brel = (const float*)d_in[10];
  const float* Wgate = (const float*)d_in[11];
  const float* bgate = (const float*)d_in[12];
  const float* Wih = (const float*)d_in[13];
  const float* bih = (const float*)d_in[14];
  const float* Whh = (const float*)d_in[15];
  const float* bhh = (const float*)d_in[16];
  const float* W1 = (const float*)d_in[17];
  const float* b1 = (const float*)d_in[18];
  const float* W2 = (const float*)d_in[19];
  const float* b2 = (const float*)d_in[20];
  const float* W3 = (const float*)d_in[21];
  const float* b3 = (const float*)d_in[22];
  const int* eo_src = (const int*)d_in[23];
  const int* eo_dst = (const int*)d_in[24];
  const int* co_src = (const int*)d_in[25];
  const int* co_dst = (const int*)d_in[26];
  const int* enroll_idx = (const int*)d_in[27];
  const int* course_idx = (const int*)d_in[28];
  float* out = (float*)d_out;

  const int H = in_sizes[4];                 // 128
  const int FE = 64, FO = 64, FC = 32;
  const int NE = in_sizes[0] / (3 * FE);     // 100000
  const int NO = in_sizes[1] / (3 * FO);     // 50000
  const int NC = in_sizes[2] / (3 * FC);     // 2000
  const int EEO = in_sizes[23];              // 800000
  const int ECO = in_sizes[25];              // 200000
  const int M = in_sizes[27];                // 4096
  const int F = 4 * H;                       // 512

  // ---- workspace bump allocator (256B aligned chunks) ----
  char* wp = (char*)d_ws;
  auto alloc = [&](size_t nfloat) -> float* {
    float* p = (float*)wp;
    wp += ((nfloat * sizeof(float) + 255) / 256) * 256;
    return p;
  };
  float* n_e    = alloc(NE);
  float* n_o_eo = alloc(NO);
  float* n_c    = alloc(NC);
  float* n_o_co = alloc(NO);
  float* st_e[2][2], *st_o[2][2], *st_c[2][2];
  for (int l = 0; l < 2; ++l)
    for (int b = 0; b < 2; ++b) {
      st_e[l][b] = alloc((size_t)NE * H);
      st_o[l][b] = alloc((size_t)NO * H);
      st_c[l][b] = alloc((size_t)NC * H);
    }
  float* proj_e = alloc((size_t)NE * H);
  float* proj_o = alloc((size_t)NO * H);
  float* proj_c = alloc((size_t)NC * H);
  float* sp_e = alloc((size_t)NE * H);
  float* sp_o = alloc((size_t)NO * H);
  float* sp_c = alloc((size_t)NC * H);
  float* agg  = alloc((size_t)NE * H);       // max rows
  float* fusedbuf = alloc((size_t)NE * H);   // gate pre-act / fused state
  float* gi = alloc((size_t)NE * 3 * H);
  float* gh = alloc((size_t)NE * 3 * H);
  float* rep = alloc((size_t)M * F);
  float* x1  = alloc((size_t)M * F);
  float* x2  = alloc((size_t)M * (F / 2));

  auto zero = [&](float* p, size_t n) {
    int blocks = (int)((n + 255) / 256);
    if (blocks > 65535) blocks = 65535;
    zero_kernel<<<blocks, 256, 0, stream>>>(p, n);
  };
  // template dispatch over (bias, rowscale, relu, addc) combos actually used
  auto gemm = [&](const float* A, const float* W, const float* bias, const float* rs,
                  float* C, int m, int k, int n, int relu, int addc) {
    dim3 grid(m / 16, n / 64);
    dim3 blk(32, 4);
    if (bias) {
      if (rs) {
        if (addc) gemm_bf16_wmma<true, true, false, true><<<grid, blk, 0, stream>>>(A, W, bias, rs, C, m, k, n);
        else      gemm_bf16_wmma<true, true, false, false><<<grid, blk, 0, stream>>>(A, W, bias, rs, C, m, k, n);
      } else {
        if (relu)      gemm_bf16_wmma<true, false, true, false><<<grid, blk, 0, stream>>>(A, W, bias, rs, C, m, k, n);
        else if (addc) gemm_bf16_wmma<true, false, false, true><<<grid, blk, 0, stream>>>(A, W, bias, rs, C, m, k, n);
        else           gemm_bf16_wmma<true, false, false, false><<<grid, blk, 0, stream>>>(A, W, bias, rs, C, m, k, n);
      }
    } else {
      if (addc) gemm_bf16_wmma<false, false, false, true><<<grid, blk, 0, stream>>>(A, W, bias, rs, C, m, k, n);
      else      gemm_bf16_wmma<false, false, false, false><<<grid, blk, 0, stream>>>(A, W, bias, rs, C, m, k, n);
    }
  };
  auto scatter = [&](const float* h, const float* ns, const int* s, const int* d,
                     float* ag, int E) {
    long threads = (long)E * 32;
    scatter_kernel<<<(int)((threads + 255) / 256), 256, 0, stream>>>(h, ns, s, d, ag, E, H);
  };

  // ---- degree norms ----
  zero(n_e, NE); zero(n_o_eo, NO); zero(n_c, NC); zero(n_o_co, NO);
  count_kernel<<<(EEO + 255) / 256, 256, 0, stream>>>(eo_src, n_e, EEO);
  count_kernel<<<(EEO + 255) / 256, 256, 0, stream>>>(eo_dst, n_o_eo, EEO);
  count_kernel<<<(ECO + 255) / 256, 256, 0, stream>>>(co_src, n_c, ECO);
  count_kernel<<<(ECO + 255) / 256, 256, 0, stream>>>(co_dst, n_o_co, ECO);
  norm_fin_kernel<<<(NE + 255) / 256, 256, 0, stream>>>(n_e, NE);
  norm_fin_kernel<<<(NO + 255) / 256, 256, 0, stream>>>(n_o_eo, NO);
  norm_fin_kernel<<<(NC + 255) / 256, 256, 0, stream>>>(n_c, NC);
  norm_fin_kernel<<<(NO + 255) / 256, 256, 0, stream>>>(n_o_co, NO);

  // ---- initial hidden states (t1 at t=0) = 0 ----
  for (int l = 0; l < 2; ++l) {
    zero(st_e[l][0], (size_t)NE * H);
    zero(st_o[l][0], (size_t)NO * H);
    zero(st_c[l][0], (size_t)NC * H);
  }

  int i_t1 = 0, i_t2 = 0;
  for (int t = 0; t < 3; ++t) {
    int i_new = (t == 0) ? 1 : i_t2;

    // projections + ReLU
    gemm(feat_e + (size_t)t * NE * FE, proj_e_w, proj_e_b, nullptr, proj_e, NE, FE, H, 1, 0);
    gemm(feat_o + (size_t)t * NO * FO, proj_o_w, proj_o_b, nullptr, proj_o, NO, FO, H, 1, 0);
    gemm(feat_c + (size_t)t * NC * FC, proj_c_w, proj_c_b, nullptr, proj_c, NC, FC, H, 1, 0);

    const float* hd_e = proj_e;
    const float* hd_o = proj_o;
    const float* hd_c = proj_c;

    for (int l = 0; l < 2; ++l) {
      const float* Wl = Wrel + (size_t)l * 4 * H * H;
      const float* bl = brel + (size_t)l * 4 * H;

      // sp_o = gconv(c->o, W0) + gconv(e->o, W3)
      zero(agg, (size_t)NO * H);
      scatter(hd_c, n_c, co_src, co_dst, agg, ECO);
      gemm(agg, Wl + 0 * (size_t)H * H, bl + 0 * H, n_o_co, sp_o, NO, H, H, 0, 0);
      zero(agg, (size_t)NO * H);
      scatter(hd_e, n_e, eo_src, eo_dst, agg, EEO);
      gemm(agg, Wl + 3 * (size_t)H * H, bl + 3 * H, n_o_eo, sp_o, NO, H, H, 0, 1);
      // sp_c = gconv(o->c, W1)
      zero(agg, (size_t)NC * H);
      scatter(hd_o, n_o_co, co_dst, co_src, agg, ECO);
      gemm(agg, Wl + 1 * (size_t)H * H, bl + 1 * H, n_c, sp_c, NC, H, H, 0, 0);
      // sp_e = gconv(o->e, W2)
      zero(agg, (size_t)NE * H);
      scatter(hd_o, n_o_eo, eo_dst, eo_src, agg, EEO);
      gemm(agg, Wl + 2 * (size_t)H * H, bl + 2 * H, n_e, sp_e, NE, H, H, 0, 0);

      const float* Wg = Wgate + (size_t)l * 3 * H * H;
      const float* bg = bgate + (size_t)l * H;
      const float* Wi = Wih + (size_t)l * H * 3 * H;
      const float* bi = bih + (size_t)l * 3 * H;
      const float* Wh = Whh + (size_t)l * H * 3 * H;
      const float* bh = bhh + (size_t)l * 3 * H;

      struct { const float* sp; float* p1; float* p2; float* outb; int n; } ty[3] = {
        { sp_e, st_e[l][i_t1], st_e[l][i_t2], st_e[l][i_new], NE },
        { sp_o, st_o[l][i_t1], st_o[l][i_t2], st_o[l][i_new], NO },
        { sp_c, st_c[l][i_t1], st_c[l][i_t2], st_c[l][i_new], NC },
      };
      for (int k = 0; k < 3; ++k) {
        int n = ty[k].n;
        const float* hfuse;
        if (t == 0) {
          hfuse = ty[k].p1;   // zeros
        } else {
          // gate pre-act = [sp|p1|p2] @ Wgate + bgate   (3 accumulating GEMMs)
          gemm(ty[k].sp, Wg + 0 * (size_t)H * H, bg, nullptr, fusedbuf, n, H, H, 0, 0);
          gemm(ty[k].p1, Wg + 1 * (size_t)H * H, nullptr, nullptr, fusedbuf, n, H, H, 0, 1);
          gemm(ty[k].p2, Wg + 2 * (size_t)H * H, nullptr, nullptr, fusedbuf, n, H, H, 0, 1);
          size_t tot = (size_t)n * H;
          gatefuse_kernel<<<(int)((tot + 255) / 256), 256, 0, stream>>>(
              fusedbuf, ty[k].p1, ty[k].p2, tot);
          hfuse = fusedbuf;
        }
        gemm(ty[k].sp, Wi, bi, nullptr, gi, n, H, 3 * H, 0, 0);
        gemm(hfuse,    Wh, bh, nullptr, gh, n, H, 3 * H, 0, 0);
        size_t tot = (size_t)n * H;
        gru_kernel<<<(int)((tot + 255) / 256), 256, 0, stream>>>(
            gi, gh, hfuse, ty[k].outb, n, H);
      }
      hd_e = st_e[l][i_new];
      hd_o = st_o[l][i_new];
      hd_c = st_c[l][i_new];
    }
    i_t2 = i_t1;
    i_t1 = i_new;
  }

  // ---- readout head ----
  {
    int tot = M * 4 * H;
    gather_rep_kernel<<<(tot + 255) / 256, 256, 0, stream>>>(
        st_e[0][i_t1], st_e[1][i_t1], st_c[0][i_t1], st_c[1][i_t1],
        enroll_idx, course_idx, rep, M, H);
  }
  gemm(rep, W1, b1, nullptr, x1, M, F, F, 1, 0);
  gemm(x1,  W2, b2, nullptr, x2, M, F, F / 2, 1, 0);
  head_kernel<<<(M + 255) / 256, 256, 0, stream>>>(x2, W3, b3, out, M, F / 2);
}
